// dayGNNnet_58342835749565
// MI455X (gfx1250) — matrix operand order repaired and checked by reading, not compiled, not run on hardware
//
#include <hip/hip_runtime.h>
#include <hip/hip_bf16.h>

// ---------------------------------------------------------------------------
// Types for CDNA5 WMMA (wave32): v_wmma_f32_16x16x32_bf16
// ---------------------------------------------------------------------------
typedef __attribute__((ext_vector_type(16))) __bf16 v16bf;
typedef __attribute__((ext_vector_type(8)))  float  v8f;

#define N_NODES 20000
#define N_EDGES 640000

// ---------------------------------------------------------------------------
// Utility kernels
// ---------------------------------------------------------------------------
__global__ void k_zero(float* __restrict__ p, size_t n) {
  size_t i = (size_t)blockIdx.x * blockDim.x + threadIdx.x;
  size_t stride = (size_t)gridDim.x * blockDim.x;
  for (; i < n; i += stride) p[i] = 0.0f;
}

__global__ void k_deg(const long long* __restrict__ dst, float* __restrict__ deg, int E) {
  int t = blockIdx.x * blockDim.x + threadIdx.x;
  if (t < E) atomicAdd(&deg[(int)dst[t]], 1.0f);
}

__global__ void k_recip(const float* __restrict__ deg, float* __restrict__ rdeg, int n) {
  int i = blockIdx.x * blockDim.x + threadIdx.x;
  if (i < n) rdeg[i] = 1.0f / fmaxf(deg[i], 1.0f);
}

// Scatter-add of node features along edges: agg[dst] += X[src].
// Memory/atomic bound -> plain global_atomic_add_f32, 4 features per thread.
__global__ void k_scatter(const float* __restrict__ X, const long long* __restrict__ src,
                          const long long* __restrict__ dst, float* __restrict__ agg,
                          int E, int d) {
  long long t = (long long)blockIdx.x * blockDim.x + threadIdx.x;
  int per = d >> 2;
  if (t >= (long long)E * per) return;
  int e = (int)(t / per);
  int c = (int)(t - (long long)e * per) << 2;
  int s  = (int)src[e];
  int dd = (int)dst[e];
  const float* xp = X + (size_t)s * d + c;
  float* ap = agg + (size_t)dd * d + c;
  atomicAdd(ap + 0, xp[0]);
  atomicAdd(ap + 1, xp[1]);
  atomicAdd(ap + 2, xp[2]);
  atomicAdd(ap + 3, xp[3]);
}

// ---------------------------------------------------------------------------
// Generic WMMA GEMM:  C[M,Nf] = (rowScale .* A)[M,K] @ W[Nf,K]^T
//                              + A2[M,K] @ W2[Nf,K]^T  (optional)
//                              + bias[Nf]              (optional)
// Block: BDIM/32 waves; each wave owns one 16x16 output tile (Nf = BDIM/2).
// The 16x32 fp32 A tile is staged in LDS via the CDNA5 async global->LDS DMA
// path (global_load_async_to_lds_b128, drained with s_wait_asynccnt), then
// converted to bf16 while packing fragments in contiguous k order (k = 16*(lane/16) + e). BDIM is a
// template parameter so the DMA-issue loop is a fully unrolled constant-trip
// sequence. rowScale (1/deg mean normalization) is row-invariant -> applied at
// pack time.
// ---------------------------------------------------------------------------
template <int BDIM>
__global__ void __launch_bounds__(BDIM)
k_gemm_wmma(const float* __restrict__ A, const float* __restrict__ W,
            const float* __restrict__ bias,
            const float* __restrict__ A2, const float* __restrict__ W2,
            const float* __restrict__ rowScale,
            float* __restrict__ C, int M, int K, int Nf) {
  __shared__ float Asf[16 * 32];          // only __shared__ object -> LDS offset 0
  constexpr int NQ = 128 / BDIM;          // float4s per thread for the 16x32 tile
  const int lane = threadIdx.x & 31;
  const int row0 = blockIdx.x * 16;
  const int n0 = (threadIdx.x >> 5) * 16;
  const int ar = lane & 15;               // A row within tile

  // Per-thread DMA slots (fixed across the K loop)
  int qr[NQ], qc4[NQ];
#pragma unroll
  for (int j = 0; j < NQ; ++j) {
    int q = threadIdx.x + j * BDIM;
    qr[j] = q >> 3;
    qc4[j] = q & 7;
  }

  v8f acc = {};

  for (int pass = 0; pass < 2; ++pass) {
    const float* Ap = pass ? A2 : A;
    const float* Wp = pass ? W2 : W;
    if (!Ap) break;  // uniform across block
    const float scale = (pass == 0 && rowScale) ? rowScale[row0 + ar] : 1.0f;

    // Hoisted per-thread row bases for the DMA issues
    const float* rowbase[NQ];
#pragma unroll
    for (int j = 0; j < NQ; ++j)
      rowbase[j] = Ap + (size_t)(row0 + qr[j]) * K + qc4[j] * 4;

    for (int kb = 0; kb < K; kb += 32) {
      // Async DMA: 16x32 fp32 tile (128 float4s) straight into LDS.
#pragma unroll
      for (int j = 0; j < NQ; ++j) {
        unsigned ldsoff = (unsigned)((threadIdx.x + j * BDIM) * 16);
        unsigned long long gaddr = (unsigned long long)(const void*)(rowbase[j] + kb);
        asm volatile("global_load_async_to_lds_b128 %0, %1, off"
                     :: "v"(ldsoff), "v"(gaddr) : "memory");
      }
      asm volatile("s_wait_asynccnt 0" ::: "memory");
      __syncthreads();

      // Pack fragments in contiguous k order, k = 16*(lane/16) + e (bf16 convert + mean scale here).
      v16bf afrag, bfrag;
      const int bc = n0 + (lane & 15);    // B column (output feature)
#pragma unroll
      for (int e = 0; e < 16; ++e) {
        int vg = e >> 1, sub = e & 1;
        // A 16x32 bf16: lanes 0-15 hold K0..15, lanes 16-31 hold K16..31
        int ak = ((lane >> 4) << 4) + e;
        afrag[e] = (__bf16)(Asf[ar * 32 + ak] * scale);
        // B 32x16 bf16: lanes 0-15 hold K0..15, lanes 16-31 hold K16..31
        int bk = ((lane >> 4) << 4) + e;
        bfrag[e] = (__bf16)Wp[(size_t)bc * K + kb + bk];
      }
      acc = __builtin_amdgcn_wmma_f32_16x16x32_bf16(
          false, afrag, false, bfrag, (short)0, acc, false, false);
      __syncthreads();
    }
  }

  // C/D layout: VGPR v, lanes 0-15 -> M=v, lanes 16-31 -> M=8+v; N = lane&15.
  const int col = n0 + (lane & 15);
  const float bv = bias ? bias[col] : 0.0f;
#pragma unroll
  for (int v = 0; v < 8; ++v) {
    int row = row0 + ((lane >> 4) << 3) + v;
    C[(size_t)row * Nf + col] = acc[v] + bv;
  }
}

// ---------------------------------------------------------------------------
// conv1: 1 -> 64 channels, 3x3 pad 1 on [H,64] "image". K=9 is too small for
// WMMA; direct VALU with weights in LDS. Padding handled branch-free: loads
// use clamped (always-legal) addresses, a cndmask select zeroes OOB taps.
// ---------------------------------------------------------------------------
__global__ void k_conv1(const float* __restrict__ I, const float* __restrict__ Wc,
                        const float* __restrict__ bias, float* __restrict__ O, int H) {
  __shared__ float w[64 * 9];
  for (int i = threadIdx.x; i < 64 * 9; i += blockDim.x) w[i] = Wc[i];
  __syncthreads();
  int p = blockIdx.x * blockDim.x + threadIdx.x;
  if (p >= H * 64) return;
  int y = p >> 6, x = p & 63;
  float patch[9];
#pragma unroll
  for (int t = 0; t < 9; ++t) {
    int yy = y + t / 3 - 1, xx = x + t % 3 - 1;   // compile-time dy/dx
    int yc = min(max(yy, 0), H - 1);
    int xc = min(max(xx, 0), 63);
    bool inb = ((unsigned)yy < (unsigned)H) & ((unsigned)xx < 64u);
    float v = I[yc * 64 + xc];
    patch[t] = inb ? v : 0.0f;
  }
  for (int co = 0; co < 64; ++co) {
    float s = bias[co];
#pragma unroll
    for (int t = 0; t < 9; ++t) s += patch[t] * w[co * 9 + t];
    O[(size_t)co * H * 64 + p] = s;
  }
}

// ---------------------------------------------------------------------------
// conv2: 64 -> 64 channels, 3x3 pad 1 — the dominant 94 GFLOP. Implicit-GEMM
// with WMMA: M = 16 pixels (same image row), N = 16 out-channels (4 waves cover
// all 64), K = 64ci*9taps = 576 in 18 bf16 WMMA steps. The im2col fill keeps
// (ci,tap) incrementally updated (k advances by exactly 32: tap+=5, ci+=3,
// carry). tap/3 uses the exact small-range identity (tap*11)>>5 (tap in 0..8),
// avoiding the generic division idiom; padding is branch-free via clamped
// addresses + one select per element (no exec save/restore in the hot loop).
// ---------------------------------------------------------------------------
__global__ void __launch_bounds__(128)
k_conv2_wmma(const float* __restrict__ I, const float* __restrict__ Wc,
             const float* __restrict__ bias, float* __restrict__ O, int H) {
  __shared__ __bf16 As[16 * 32];
  const int lane = threadIdx.x & 31;
  const int wave = threadIdx.x >> 5;
  const int p0 = blockIdx.x * 16;        // 16 consecutive pixels; 64%16==0 -> same y
  const int y0 = p0 >> 6;
  const int x0 = p0 & 63;
  const int co0 = wave * 16;
  const int Ktot = 576;                   // k = ci*9 + tap (matches OIHW flat layout)
  v8f acc = {};

  // Per-thread fill slots: blockDim=128 -> 4 LDS elements per thread, fixed (r,c).
  int sr[4], sci[4], stap[4];
#pragma unroll
  for (int j = 0; j < 4; ++j) {
    int idx = threadIdx.x + j * 128;
    sr[j] = idx >> 5;
    int c = idx & 31;
    sci[j] = c / 9;                       // once, outside the K loop
    stap[j] = c - sci[j] * 9;
  }

  for (int kb = 0; kb < Ktot; kb += 32) {
#pragma unroll
    for (int j = 0; j < 4; ++j) {
      int ci = sci[j], tap = stap[j];
      int d3 = (tap * 11) >> 5;           // == tap/3 for tap in [0,8]
      int y = y0 + d3 - 1;
      int x = x0 + sr[j] + (tap - 3 * d3) - 1;
      int yc = min(max(y, 0), H - 1);
      int xc = min(max(x, 0), 63);
      bool inb = ((unsigned)y < (unsigned)H) & ((unsigned)x < 64u);
      float v = I[((size_t)ci * H + yc) * 64 + xc];   // always-legal address
      As[threadIdx.x + j * 128] = (__bf16)(inb ? v : 0.0f);
      // advance this slot's k by 32: 32 = 3*9 + 5
      tap += 5; ci += 3;
      if (tap >= 9) { tap -= 9; ++ci; }
      sci[j] = ci; stap[j] = tap;
    }
    __syncthreads();

    v16bf afrag, bfrag;
    const int ar = lane & 15;
    const int co = co0 + (lane & 15);
#pragma unroll
    for (int e = 0; e < 16; ++e) {
      int vg = e >> 1, sub = e & 1;
      int ak = ((lane >> 4) << 4) + e;
      afrag[e] = As[ar * 32 + ak];
      int bk = kb + ((lane >> 4) << 4) + e;
      bfrag[e] = (__bf16)Wc[(size_t)co * Ktot + bk];
    }
    acc = __builtin_amdgcn_wmma_f32_16x16x32_bf16(
        false, afrag, false, bfrag, (short)0, acc, false, false);
    __syncthreads();
  }

  const int co = co0 + (lane & 15);
  const float bv = bias[co];
#pragma unroll
  for (int v = 0; v < 8; ++v) {
    int p = p0 + ((lane >> 4) << 3) + v;
    O[(size_t)co * H * 64 + p] = acc[v] + bv;
  }
}

// ---------------------------------------------------------------------------
// conv3: 64 -> 1 channel, 3x3 pad 1, fused with residual add (out_1).
// Channel reduction -> direct VALU. Per-tap clamped offsets and validity are
// hoisted out of the 64-channel loop; the inner loop is load+select+FMA only.
// ---------------------------------------------------------------------------
__global__ void k_conv3_res(const float* __restrict__ I, const float* __restrict__ Wc,
                            const float* __restrict__ bias, const float* __restrict__ res,
                            float* __restrict__ O, int H) {
  __shared__ float w[64 * 9];
  for (int i = threadIdx.x; i < 64 * 9; i += blockDim.x) w[i] = Wc[i];
  __syncthreads();
  int p = blockIdx.x * blockDim.x + threadIdx.x;
  if (p >= H * 64) return;
  int y = p >> 6, x = p & 63;

  int off[9];
  bool msk[9];
#pragma unroll
  for (int t = 0; t < 9; ++t) {
    int yy = y + t / 3 - 1, xx = x + t % 3 - 1;   // compile-time dy/dx
    int yc = min(max(yy, 0), H - 1);
    int xc = min(max(xx, 0), 63);
    off[t] = yc * 64 + xc;
    msk[t] = ((unsigned)yy < (unsigned)H) & ((unsigned)xx < 64u);
  }

  float s = bias[0];
  for (int ci = 0; ci < 64; ++ci) {
    const float* Ic = I + (size_t)ci * H * 64;
    const float* wc = w + ci * 9;
#pragma unroll
    for (int t = 0; t < 9; ++t) {
      float v = Ic[off[t]];
      s = fmaf(msk[t] ? v : 0.0f, wc[t], s);
    }
  }
  O[p] = s + res[p];
}

// ---------------------------------------------------------------------------
// Orchestration
// ---------------------------------------------------------------------------
extern "C" void kernel_launch(void* const* d_in, const int* in_sizes, int n_in,
                              void* d_out, int out_size, void* d_ws, size_t ws_size,
                              hipStream_t stream) {
  (void)in_sizes; (void)n_in; (void)out_size; (void)ws_size;
  const int N = N_NODES, E = N_EDGES;

  const float*     x   = (const float*)d_in[0];
  const long long* ei  = (const long long*)d_in[1];   // jnp.int64
  const long long* src = ei;
  const long long* dst = ei + E;
  const float* Wl1 = (const float*)d_in[2];
  const float* bl1 = (const float*)d_in[3];
  const float* Wr1 = (const float*)d_in[4];
  const float* Wl2 = (const float*)d_in[5];
  const float* bl2 = (const float*)d_in[6];
  const float* Wr2 = (const float*)d_in[7];
  const float* Wl3 = (const float*)d_in[8];
  const float* bl3 = (const float*)d_in[9];
  const float* Wr3 = (const float*)d_in[10];
  const float* Wc1 = (const float*)d_in[11];
  const float* bc1 = (const float*)d_in[12];
  const float* Wc2 = (const float*)d_in[13];
  const float* bc2 = (const float*)d_in[14];
  const float* Wc3 = (const float*)d_in[15];
  const float* bc3 = (const float*)d_in[16];
  const float* W2  = (const float*)d_in[17];
  const float* b2  = (const float*)d_in[18];
  float* out = (float*)d_out;

  // Workspace layout (floats)
  float* ws    = (float*)d_ws;
  float* deg   = ws;                               // N
  float* rdeg  = deg + N;                          // N
  float* agg   = rdeg + N;                         // N*128 (max d_in)
  float* hA    = agg + (size_t)N * 128;            // N*64
  float* hB    = hA + (size_t)N * 64;              // N*64
  float* hC    = hB + (size_t)N * 64;              // N*64
  float* convA = hC + (size_t)N * 64;              // 64*N*64
  float* convB = convA + (size_t)64 * N * 64;      // 64*N*64

  const int pix = N * 64;

  // Degrees once (edges constant across all sage layers)
  k_zero<<<2048, 256, 0, stream>>>(deg, (size_t)N);
  k_deg<<<(E + 255) / 256, 256, 0, stream>>>(dst, deg, E);
  k_recip<<<(N + 255) / 256, 256, 0, stream>>>(deg, rdeg, N);

  auto sage64 = [&](const float* xin, int din, const float* Wl, const float* bl,
                    const float* Wr, float* o) {
    k_zero<<<2048, 256, 0, stream>>>(agg, (size_t)N * din);
    long long tot = (long long)E * (din >> 2);
    k_scatter<<<(unsigned)((tot + 255) / 256), 256, 0, stream>>>(xin, src, dst, agg, E, din);
    k_gemm_wmma<128><<<N / 16, 128, 0, stream>>>(agg, Wl, bl, xin, Wr, rdeg, o, N, din, 64);
  };

  // h = sage1(x); h = sage2(h); h = sage2(h)  -> out_1 in hA
  sage64(x,  128, Wl1, bl1, Wr1, hB);
  sage64(hB, 64,  Wl2, bl2, Wr2, hC);
  sage64(hC, 64,  Wl2, bl2, Wr2, hA);

  // Conv stack on [1,1,N,64] image; out_1 = hA kept for the residual
  k_conv1<<<(pix + 255) / 256, 256, 0, stream>>>(hA, Wc1, bc1, convA, N);
  k_conv2_wmma<<<pix / 16, 128, 0, stream>>>(convA, Wc2, bc2, convB, N);
  k_conv3_res<<<(pix + 255) / 256, 256, 0, stream>>>(convB, Wc3, bc3, hA, hB, N);

  // out = out @ W2^T + b2
  k_gemm_wmma<128><<<N / 16, 128, 0, stream>>>(hB, W2, b2, nullptr, nullptr, nullptr,
                                               hC, N, 64, 64);

  // h = sage2(out); h = sage2(h); h = sage3(h, Wl3/Wr3 -> 32 features)
  sage64(hC, 64, Wl2, bl2, Wr2, hA);
  sage64(hA, 64, Wl2, bl2, Wr2, hB);
  {
    k_zero<<<2048, 256, 0, stream>>>(agg, (size_t)N * 64);
    long long tot = (long long)E * (64 >> 2);
    k_scatter<<<(unsigned)((tot + 255) / 256), 256, 0, stream>>>(hB, src, dst, agg, E, 64);
    k_gemm_wmma<64><<<N / 16, 64, 0, stream>>>(agg, Wl3, bl3, hB, Wr3, rdeg, out, N, 64, 32);
  }
}
